// HybridDeformableAttentionBlock_18433999634675
// MI455X (gfx1250) — matrix operand order repaired and checked by reading, not compiled, not run on hardware
//
#include <hip/hip_runtime.h>
#include <hip/hip_bf16.h>

typedef __bf16 bf16_t;
typedef __bf16 v16bf __attribute__((ext_vector_type(16)));
typedef float  v8f   __attribute__((ext_vector_type(8)));

union Frag { uint4 u[2]; v16bf v; };

__device__ __forceinline__ bf16_t f2bf(float f) {
  unsigned u = __builtin_bit_cast(unsigned, f);
  unsigned r = u + 0x7FFFu + ((u >> 16) & 1u);   // round-to-nearest-even
  unsigned short h = (unsigned short)(r >> 16);
  return __builtin_bit_cast(bf16_t, h);
}

// ---------------------------------------------------------------------------
// WMMA GEMM: OUT[M][N] = A[M][K] * Bt[N][K]^T (+bias, +resid, act)
// One wave computes a 16x64 strip (4 N-tiles), software-pipelined: fragments
// for k-step i+1 are loaded before the 4 v_wmma ops of step i are issued,
// so the matrix pipe overlaps ~10 in-flight global_load_b128.
// act: 0 = none, 1 = exact GELU
// ---------------------------------------------------------------------------
__global__ __launch_bounds__(32)
void gemm16(const bf16_t* __restrict__ A, const bf16_t* __restrict__ Bt,
            const float* __restrict__ bias, const float* __restrict__ resid,
            float* __restrict__ outF, bf16_t* __restrict__ outB,
            int M, int K, int N, int act)
{
  int n0 = blockIdx.x * 64, m0 = blockIdx.y * 16;
  int lane = threadIdx.x;
  int g = lane >> 4, col = lane & 15;
  int abase = g * 16;
  const bf16_t* arow  = A  + (size_t)(m0 + col) * K;
  const bf16_t* brow0 = Bt + (size_t)(n0 + col) * K + g * 16;
  const size_t  bstr  = (size_t)16 * K;          // N-tile stride in Bt rows
  const v8f vz = {0.f,0.f,0.f,0.f,0.f,0.f,0.f,0.f};
  v8f acc[4] = {vz, vz, vz, vz};

  auto loadA = [&](Frag& a, int k0) {
    a.u[0] = *(const uint4*)(arow + k0 + abase);
    a.u[1] = *(const uint4*)(arow + k0 + 8 + abase);
  };
  auto loadB = [&](Frag (&b)[4], int k0) {
    #pragma unroll
    for (int t = 0; t < 4; ++t) {
      const bf16_t* br = brow0 + (size_t)t * bstr + k0;
      b[t].u[0] = *(const uint4*)(br);
      b[t].u[1] = *(const uint4*)(br + 8);
    }
  };

  Frag a0, a1, b0[4], b1[4];
  loadA(a0, 0); loadB(b0, 0);
  int k0 = 0;
  for (; k0 + 64 <= K; k0 += 64) {
    loadA(a1, k0 + 32); loadB(b1, k0 + 32);
    if (k0 + 192 < K) {                          // global_prefetch_b8 (gfx1250)
      __builtin_prefetch(arow + k0 + 192, 0, 1);
      __builtin_prefetch(brow0 + k0 + 192, 0, 1);
    }
    #pragma unroll
    for (int t = 0; t < 4; ++t)
      acc[t] = __builtin_amdgcn_wmma_f32_16x16x32_bf16(false, a0.v, false, b0[t].v,
                                                       (short)0, acc[t], false, false);
    if (k0 + 96 <= K) { loadA(a0, k0 + 64); loadB(b0, k0 + 64); }
    #pragma unroll
    for (int t = 0; t < 4; ++t)
      acc[t] = __builtin_amdgcn_wmma_f32_16x16x32_bf16(false, a1.v, false, b1[t].v,
                                                       (short)0, acc[t], false, false);
  }
  for (; k0 < K; k0 += 32) {                     // generic tail (dead when K%64==0)
    loadA(a0, k0); loadB(b0, k0);
    #pragma unroll
    for (int t = 0; t < 4; ++t)
      acc[t] = __builtin_amdgcn_wmma_f32_16x16x32_bf16(false, a0.v, false, b0[t].v,
                                                       (short)0, acc[t], false, false);
  }

  #pragma unroll
  for (int t = 0; t < 4; ++t) {
    int nn = n0 + t * 16 + col;
    float bv = bias ? bias[nn] : 0.f;
    for (int r = 0; r < 8; ++r) {
      int mm = m0 + r + 8 * g;
      float v = acc[t][r] + bv;
      if (resid) v += resid[(size_t)mm * N + nn];
      if (act == 1) v = 0.5f * v * (1.f + erff(v * 0.70710678118654752f));
      if (outF) outF[(size_t)mm * N + nn] = v;
      if (outB) outB[(size_t)mm * N + nn] = f2bf(v);
    }
  }
}

// ---------------------------------------------------------------------------
// LayerNorm over last dim C; writes f32 (optional) and bf16 outputs.
// ---------------------------------------------------------------------------
__global__ __launch_bounds__(256)
void layernorm_k(const float* __restrict__ in, const float* __restrict__ s,
                 const float* __restrict__ bi, float* __restrict__ outF,
                 bf16_t* __restrict__ outB, int C)
{
  __shared__ float red[256];
  __shared__ float red2[256];
  int row = blockIdx.x, tid = threadIdx.x;
  const float* r = in + (size_t)row * C;
  float sum = 0.f, sq = 0.f;
  for (int c = tid; c < C; c += 256) { float v = r[c]; sum += v; sq += v * v; }
  red[tid] = sum; red2[tid] = sq; __syncthreads();
  for (int off = 128; off > 0; off >>= 1) {
    if (tid < off) { red[tid] += red[tid + off]; red2[tid] += red2[tid + off]; }
    __syncthreads();
  }
  float mean = red[0] / C;
  float var  = red2[0] / C - mean * mean;
  float inv  = rsqrtf(var + 1e-6f);
  for (int c = tid; c < C; c += 256) {
    float v = (r[c] - mean) * inv * s[c] + bi[c];
    if (outF) outF[(size_t)row * C + c] = v;
    outB[(size_t)row * C + c] = f2bf(v);
  }
}

// Wt[n*K + k] = bf16(W[k*N + n])
__global__ void tconv_k(const float* __restrict__ W, bf16_t* __restrict__ Wt,
                        int K, int N)
{
  size_t total = (size_t)K * N;
  for (size_t i = (size_t)blockIdx.x * 256 + threadIdx.x; i < total;
       i += (size_t)gridDim.x * 256) {
    size_t n = i / K, k = i % K;
    Wt[i] = f2bf(W[k * (size_t)N + n]);
  }
}

__global__ void conv_k(const float* __restrict__ W, bf16_t* __restrict__ Wt,
                       size_t total)
{
  for (size_t i = (size_t)blockIdx.x * 256 + threadIdx.x; i < total;
       i += (size_t)gridDim.x * 256)
    Wt[i] = f2bf(W[i]);
}

// V [B,N,C] bf16 -> VT [B,NH,64,N] bf16
__global__ void vtrans_k(const bf16_t* __restrict__ V, bf16_t* __restrict__ VT)
{
  const size_t total = (size_t)2 * 1024 * 768;
  for (size_t i = (size_t)blockIdx.x * 256 + threadIdx.x; i < total;
       i += (size_t)gridDim.x * 256) {
    size_t b = i / (1024 * 768), rem = i % (1024 * 768);
    size_t n = rem / 768, c = rem % 768;
    size_t h = c / 64, dd = c % 64;
    VT[(((b * 12 + h) * 64 + dd) * 1024) + n] = V[i];
  }
}

// ---------------------------------------------------------------------------
// Flash-style attention: one wave per (b, h, 16-query tile). d=64, N=1024.
// ---------------------------------------------------------------------------
__global__ __launch_bounds__(32)
void attn_k(const bf16_t* __restrict__ Q, const bf16_t* __restrict__ Km,
            const bf16_t* __restrict__ VT, bf16_t* __restrict__ ctx)
{
  const int Nn = 1024, C = 768, D = 64;
  int b = blockIdx.z, h = blockIdx.y, n0q = blockIdx.x * 16;
  int lane = threadIdx.x, g = lane >> 4, col = lane & 15;
  int abase = g * 16;
  __shared__ __align__(16) bf16_t pbuf[16][32];

  const bf16_t* qrow = Q + ((size_t)(b * Nn + n0q + col)) * C + h * D;
  Frag qa0, qa1;
  qa0.u[0] = *(const uint4*)(qrow + 0  + abase);
  qa0.u[1] = *(const uint4*)(qrow + 8 + abase);
  qa1.u[0] = *(const uint4*)(qrow + 32 + abase);
  qa1.u[1] = *(const uint4*)(qrow + 40 + abase);

  float mrow[8], lrow[8];
  const v8f vz = {0.f,0.f,0.f,0.f,0.f,0.f,0.f,0.f};
  v8f cacc[4] = {vz, vz, vz, vz};
  for (int r = 0; r < 8; ++r) { mrow[r] = -1e30f; lrow[r] = 0.f; }

  for (int key0 = 0; key0 < Nn; key0 += 32) {
    const bf16_t* krow0 = Km + ((size_t)(b * Nn + key0 + col)) * C + h * D + g * 16;
    const bf16_t* krow1 = krow0 + (size_t)16 * C;
    v8f s0 = vz, s1 = vz;
    Frag kb00, kb01, kb10, kb11;
    kb00.u[0] = *(const uint4*)(krow0 + 0);
    kb00.u[1] = *(const uint4*)(krow0 + 8);
    kb01.u[0] = *(const uint4*)(krow0 + 32);
    kb01.u[1] = *(const uint4*)(krow0 + 40);
    kb10.u[0] = *(const uint4*)(krow1 + 0);
    kb10.u[1] = *(const uint4*)(krow1 + 8);
    kb11.u[0] = *(const uint4*)(krow1 + 32);
    kb11.u[1] = *(const uint4*)(krow1 + 40);
    s0 = __builtin_amdgcn_wmma_f32_16x16x32_bf16(false, qa0.v, false, kb00.v, (short)0, s0, false, false);
    s0 = __builtin_amdgcn_wmma_f32_16x16x32_bf16(false, qa1.v, false, kb01.v, (short)0, s0, false, false);
    s1 = __builtin_amdgcn_wmma_f32_16x16x32_bf16(false, qa0.v, false, kb10.v, (short)0, s1, false, false);
    s1 = __builtin_amdgcn_wmma_f32_16x16x32_bf16(false, qa1.v, false, kb11.v, (short)0, s1, false, false);

    for (int r = 0; r < 8; ++r) {
      float a0 = s0[r] * 0.125f, a1 = s1[r] * 0.125f;
      float t = fmaxf(a0, a1);
      for (int off = 1; off < 16; off <<= 1) t = fmaxf(t, __shfl_xor(t, off, 16));
      float mnew  = fmaxf(mrow[r], t);
      float alpha = __expf(mrow[r] - mnew);
      float p0 = __expf(a0 - mnew), p1 = __expf(a1 - mnew);
      float ps = p0 + p1;
      for (int off = 1; off < 16; off <<= 1) ps += __shfl_xor(ps, off, 16);
      lrow[r] = lrow[r] * alpha + ps;
      mrow[r] = mnew;
      for (int t4 = 0; t4 < 4; ++t4) cacc[t4][r] *= alpha;
      int mm = r + 8 * g;
      pbuf[mm][col]      = f2bf(p0);
      pbuf[mm][16 + col] = f2bf(p1);
    }
    __syncthreads();
    Frag pa;
    pa.u[0] = *(const uint4*)(&pbuf[col][abase]);
    pa.u[1] = *(const uint4*)(&pbuf[col][8 + abase]);
    Frag vb[4];
    #pragma unroll
    for (int t4 = 0; t4 < 4; ++t4) {
      const bf16_t* vrow = VT + ((size_t)((b * 12 + h) * 64 + t4 * 16 + col)) * Nn
                              + key0 + g * 16;
      vb[t4].u[0] = *(const uint4*)(vrow);
      vb[t4].u[1] = *(const uint4*)(vrow + 8);
    }
    #pragma unroll
    for (int t4 = 0; t4 < 4; ++t4)
      cacc[t4] = __builtin_amdgcn_wmma_f32_16x16x32_bf16(false, pa.v, false, vb[t4].v,
                                                         (short)0, cacc[t4], false, false);
    __syncthreads();
  }
  for (int r = 0; r < 8; ++r) {
    float inv = 1.f / lrow[r];
    int mm = n0q + r + 8 * g;
    for (int t4 = 0; t4 < 4; ++t4)
      ctx[((size_t)(b * Nn + mm)) * C + h * D + t4 * 16 + col] = f2bf(cacc[t4][r] * inv);
  }
}

// ---------------------------------------------------------------------------
// 3x3 conv producing 18 offset + 9 modulator channels (sigmoid'ed).
// One block per pixel; threads split the 768 input channels.
// ---------------------------------------------------------------------------
__global__ __launch_bounds__(256)
void offconv_k(const float* __restrict__ xn, const float* __restrict__ offw,
               const float* __restrict__ offb, const float* __restrict__ modw,
               const float* __restrict__ modb, float* __restrict__ off,
               float* __restrict__ mod)
{
  const int C = 768, Hh = 32, Wd = 32, Nn = 1024;
  int pix = blockIdx.x;
  int b = pix >> 10, p = pix & 1023;
  int iy = p >> 5, ix = p & 31;
  int tid = threadIdx.x;
  float acc[27];
  for (int i = 0; i < 27; ++i) acc[i] = 0.f;
  for (int c = tid; c < C; c += 256) {
    for (int tap = 0; tap < 9; ++tap) {
      int yy = iy + tap / 3 - 1, xx = ix + tap % 3 - 1;
      if (yy < 0 || yy >= Hh || xx < 0 || xx >= Wd) continue;
      float inv = xn[((size_t)(b * Nn + yy * Wd + xx)) * C + c];
      size_t wbase = (size_t)c * 9 + tap;
      #pragma unroll
      for (int oc = 0; oc < 18; ++oc) acc[oc]      += inv * offw[(size_t)oc * C * 9 + wbase];
      #pragma unroll
      for (int oc = 0; oc < 9;  ++oc) acc[18 + oc] += inv * modw[(size_t)oc * C * 9 + wbase];
    }
  }
  __shared__ float red[256];
  for (int oc = 0; oc < 27; ++oc) {
    red[tid] = acc[oc]; __syncthreads();
    for (int o = 128; o > 0; o >>= 1) { if (tid < o) red[tid] += red[tid + o]; __syncthreads(); }
    if (tid == 0) {
      float v = red[0];
      if (oc < 18) off[((size_t)(b * 18 + oc)) * Nn + p] = v + offb[oc];
      else {
        float m = v + modb[oc - 18];
        mod[((size_t)(b * 9 + (oc - 18))) * Nn + p] = 1.f / (1.f + __expf(-m));
      }
    }
    __syncthreads();
  }
}

// ---------------------------------------------------------------------------
// Deformable bilinear sampling of the (zero-padded) value stream; writes
// feat[pix][c*9 + j] in bf16 (A operand of the dca GEMM), modulator folded in.
// ---------------------------------------------------------------------------
__global__ __launch_bounds__(256)
void dsample_k(const float* __restrict__ src, const float* __restrict__ off,
               const float* __restrict__ mod, bf16_t* __restrict__ feat)
{
  const int C = 768, Nn = 1024, Wd = 32;
  int pix = blockIdx.x, b = pix >> 10, p = pix & 1023;
  int iy = p >> 5, ix = p & 31;
  int tid = threadIdx.x;
  __shared__ int   sidx[9][4];
  __shared__ float swt[9][4];
  if (tid < 9) {
    int j = tid;
    float py = (float)iy + (float)(j / 3 - 1) + off[((size_t)(b * 18 + j)) * Nn + p];
    float px = (float)ix + (float)(j % 3 - 1) + off[((size_t)(b * 18 + 9 + j)) * Nn + p];
    const float sc = 34.0f / 33.0f;                 // Wp/(Wp-1), pad=1 -> 34
    float gx = px * sc - 0.5f, gy = py * sc - 0.5f;
    float x0f = floorf(gx), y0f = floorf(gy);
    float wx = gx - x0f, wy = gy - y0f;
    int x0 = (int)x0f, y0 = (int)y0f;
    float mj = mod[((size_t)(b * 9 + j)) * Nn + p];
    int   xs[4] = {x0, x0 + 1, x0,     x0 + 1};
    int   ys[4] = {y0, y0,     y0 + 1, y0 + 1};
    float ws[4] = {(1 - wx) * (1 - wy), wx * (1 - wy), (1 - wx) * wy, wx * wy};
    for (int q = 0; q < 4; ++q) {
      int xi = xs[q], yi = ys[q];
      bool ok = (xi >= 1 && xi <= 32 && yi >= 1 && yi <= 32);  // pad ring is zero
      sidx[j][q] = ok ? ((yi - 1) * Wd + (xi - 1)) : -1;
      swt[j][q]  = ok ? ws[q] * mj : 0.f;
    }
  }
  __syncthreads();
  for (int c = tid; c < C; c += 256) {
    bf16_t outv[9];
    for (int j = 0; j < 9; ++j) {
      float v = 0.f;
      for (int q = 0; q < 4; ++q) {
        int id = sidx[j][q];
        if (id >= 0) v += swt[j][q] * src[((size_t)(b * Nn + id)) * C + c];
      }
      outv[j] = f2bf(v);
    }
    bf16_t* dst = feat + (size_t)pix * 6912 + (size_t)c * 9;
    for (int j = 0; j < 9; ++j) dst[j] = outv[j];
  }
}

// x1 = base + g*as + (1-g)*ac  with g = sigmoid(gate[0])
__global__ void combine_k(const float* __restrict__ base, const float* __restrict__ as_,
                          const float* __restrict__ ac, const float* __restrict__ gate,
                          float* __restrict__ out, size_t total)
{
  float gv = 1.f / (1.f + __expf(-gate[0]));
  for (size_t i = (size_t)blockIdx.x * 256 + threadIdx.x; i < total;
       i += (size_t)gridDim.x * 256)
    out[i] = base[i] + gv * as_[i] + (1.f - gv) * ac[i];
}

// ---------------------------------------------------------------------------
extern "C" void kernel_launch(void* const* d_in, const int* in_sizes, int n_in,
                              void* d_out, int out_size, void* d_ws, size_t ws_size,
                              hipStream_t stream) {
  (void)in_sizes; (void)n_in; (void)out_size; (void)ws_size;
  const int B = 2, Nn = 1024, C = 768, MLPD = 3072;
  const int M = B * Nn;                 // 2048
  const size_t NC = (size_t)M * C;      // 1,572,864

  const float* x      = (const float*)d_in[0];
  const float* y      = (const float*)d_in[1];
  const float* ln1x_s = (const float*)d_in[2];
  const float* ln1x_b = (const float*)d_in[3];
  const float* ln1y_s = (const float*)d_in[4];
  const float* ln1y_b = (const float*)d_in[5];
  const float* ln2x_s = (const float*)d_in[6];
  const float* ln2x_b = (const float*)d_in[7];
  const float* ln2y_s = (const float*)d_in[8];
  const float* ln2y_b = (const float*)d_in[9];
  const float* Wmat[8] = { (const float*)d_in[10], (const float*)d_in[11],
                           (const float*)d_in[12], (const float*)d_in[13],
                           (const float*)d_in[14], (const float*)d_in[15],
                           (const float*)d_in[16], (const float*)d_in[17] };
  const float* bvec[8] = { (const float*)d_in[18], (const float*)d_in[19],
                           (const float*)d_in[20], (const float*)d_in[21],
                           (const float*)d_in[22], (const float*)d_in[23],
                           (const float*)d_in[24], (const float*)d_in[25] };
  const float* gate_x = (const float*)d_in[26];
  const float* gate_y = (const float*)d_in[27];
  const float* off_w  = (const float*)d_in[28];
  const float* off_b  = (const float*)d_in[29];
  const float* mod_w  = (const float*)d_in[30];
  const float* mod_b  = (const float*)d_in[31];
  const float* dca_w  = (const float*)d_in[32];
  const float* fc1x_w = (const float*)d_in[33];
  const float* fc1x_b = (const float*)d_in[34];
  const float* fc2x_w = (const float*)d_in[35];
  const float* fc2x_b = (const float*)d_in[36];
  const float* fc1y_w = (const float*)d_in[37];
  const float* fc1y_b = (const float*)d_in[38];
  const float* fc2y_w = (const float*)d_in[39];
  const float* fc2y_b = (const float*)d_in[40];

  // Workspace carve-out
  char* wsb = (char*)d_ws;
  size_t pos = 0;
  auto alloc = [&](size_t bytes) -> void* {
    void* p = wsb + pos;
    pos = (pos + bytes + 255) & ~(size_t)255;
    return p;
  };
  float*  xn   = (float*)alloc(NC * 4);
  float*  yn   = (float*)alloc(NC * 4);
  bf16_t* xnb  = (bf16_t*)alloc(NC * 2);
  bf16_t* ynb  = (bf16_t*)alloc(NC * 2);
  bf16_t* qxb  = (bf16_t*)alloc(NC * 2);
  bf16_t* kxb  = (bf16_t*)alloc(NC * 2);
  bf16_t* vxb  = (bf16_t*)alloc(NC * 2);
  bf16_t* qyb  = (bf16_t*)alloc(NC * 2);
  bf16_t* kyb  = (bf16_t*)alloc(NC * 2);
  bf16_t* vyb  = (bf16_t*)alloc(NC * 2);
  bf16_t* vtx  = (bf16_t*)alloc(NC * 2);
  bf16_t* vty  = (bf16_t*)alloc(NC * 2);
  bf16_t* ctxx = (bf16_t*)alloc(NC * 2);
  bf16_t* ctxy = (bf16_t*)alloc(NC * 2);
  float*  asx  = (float*)alloc(NC * 4);
  float*  asy  = (float*)alloc(NC * 4);
  float*  offb_ = (float*)alloc((size_t)B * 18 * Nn * 4);
  float*  modb_ = (float*)alloc((size_t)B * 9 * Nn * 4);
  bf16_t* feat = (bf16_t*)alloc((size_t)M * (C * 9) * 2);
  float*  acx  = (float*)alloc(NC * 4);
  float*  acy  = (float*)alloc(NC * 4);
  float*  x1   = (float*)alloc(NC * 4);
  float*  y1   = (float*)alloc(NC * 4);
  bf16_t* x1nb = (bf16_t*)alloc(NC * 2);
  bf16_t* y1nb = (bf16_t*)alloc(NC * 2);
  bf16_t* hid  = (bf16_t*)alloc((size_t)M * MLPD * 2);
  bf16_t* Wt[8];
  for (int i = 0; i < 8; ++i) Wt[i] = (bf16_t*)alloc((size_t)C * C * 2);
  bf16_t* dcab  = (bf16_t*)alloc((size_t)C * C * 9 * 2);
  bf16_t* fc1xt = (bf16_t*)alloc((size_t)C * MLPD * 2);
  bf16_t* fc2xt = (bf16_t*)alloc((size_t)C * MLPD * 2);
  bf16_t* fc1yt = (bf16_t*)alloc((size_t)C * MLPD * 2);
  bf16_t* fc2yt = (bf16_t*)alloc((size_t)C * MLPD * 2);

  float* xout = (float*)d_out;
  float* yout = xout + NC;

  auto gemm = [&](const bf16_t* A, const bf16_t* Bt, const float* bias,
                  const float* resid, float* oF, bf16_t* oB,
                  int Mm, int Kk, int Nout, int act) {
    gemm16<<<dim3(Nout / 64, Mm / 16), dim3(32), 0, stream>>>(
        A, Bt, bias, resid, oF, oB, Mm, Kk, Nout, act);
  };

  // 1. Convert + transpose weights to bf16
  for (int i = 0; i < 8; ++i)
    tconv_k<<<1024, 256, 0, stream>>>(Wmat[i], Wt[i], C, C);
  conv_k<<<2048, 256, 0, stream>>>(dca_w, dcab, (size_t)C * C * 9); // already [o][i]
  tconv_k<<<2048, 256, 0, stream>>>(fc1x_w, fc1xt, C, MLPD);
  tconv_k<<<2048, 256, 0, stream>>>(fc2x_w, fc2xt, MLPD, C);
  tconv_k<<<2048, 256, 0, stream>>>(fc1y_w, fc1yt, C, MLPD);
  tconv_k<<<2048, 256, 0, stream>>>(fc2y_w, fc2yt, MLPD, C);

  // 2. LayerNorm 1
  layernorm_k<<<M, 256, 0, stream>>>(x, ln1x_s, ln1x_b, xn, xnb, C);
  layernorm_k<<<M, 256, 0, stream>>>(y, ln1y_s, ln1y_b, yn, ynb, C);

  // 3. Q/K/V projections (bf16 outputs)
  gemm(xnb, Wt[0], bvec[0], nullptr, nullptr, qxb, M, C, C, 0);
  gemm(xnb, Wt[1], bvec[1], nullptr, nullptr, kxb, M, C, C, 0);
  gemm(xnb, Wt[2], bvec[2], nullptr, nullptr, vxb, M, C, C, 0);
  gemm(ynb, Wt[4], bvec[4], nullptr, nullptr, qyb, M, C, C, 0);
  gemm(ynb, Wt[5], bvec[5], nullptr, nullptr, kyb, M, C, C, 0);
  gemm(ynb, Wt[6], bvec[6], nullptr, nullptr, vyb, M, C, C, 0);

  // 4. V transpose for attention B operand
  vtrans_k<<<2048, 256, 0, stream>>>(vxb, vtx);
  vtrans_k<<<2048, 256, 0, stream>>>(vyb, vty);

  // 5. Attention (flash-style, WMMA)
  attn_k<<<dim3(Nn / 16, 12, B), dim3(32), 0, stream>>>(qxb, kxb, vtx, ctxx);
  attn_k<<<dim3(Nn / 16, 12, B), dim3(32), 0, stream>>>(qyb, kyb, vty, ctxy);

  // 6. Output projections
  gemm(ctxx, Wt[3], bvec[3], nullptr, asx, nullptr, M, C, C, 0);
  gemm(ctxy, Wt[7], bvec[7], nullptr, asy, nullptr, M, C, C, 0);

  // 7. Deformable cross-attention, x queries sampling y
  offconv_k<<<M, 256, 0, stream>>>(xn, off_w, off_b, mod_w, mod_b, offb_, modb_);
  dsample_k<<<M, 256, 0, stream>>>(yn, offb_, modb_, feat);
  gemm(feat, dcab, nullptr, nullptr, acx, nullptr, M, C * 9, C, 0);
  //    and y queries sampling x (buffers reused; launches are stream-ordered)
  offconv_k<<<M, 256, 0, stream>>>(yn, off_w, off_b, mod_w, mod_b, offb_, modb_);
  dsample_k<<<M, 256, 0, stream>>>(xn, offb_, modb_, feat);
  gemm(feat, dcab, nullptr, nullptr, acy, nullptr, M, C * 9, C, 0);

  // 8. Gated residual combine
  combine_k<<<2048, 256, 0, stream>>>(x, asx, acx, gate_x, x1, NC);
  combine_k<<<2048, 256, 0, stream>>>(y, asy, acy, gate_y, y1, NC);

  // 9. MLP (LN2 -> fc1+GELU -> fc2 + residual), x then y (hid reused)
  layernorm_k<<<M, 256, 0, stream>>>(x1, ln2x_s, ln2x_b, nullptr, x1nb, C);
  gemm(x1nb, fc1xt, fc1x_b, nullptr, nullptr, hid, M, C, MLPD, 1);
  gemm(hid, fc2xt, fc2x_b, x1, xout, nullptr, M, MLPD, C, 0);

  layernorm_k<<<M, 256, 0, stream>>>(y1, ln2y_s, ln2y_b, nullptr, y1nb, C);
  gemm(y1nb, fc1yt, fc1y_b, nullptr, nullptr, hid, M, C, MLPD, 1);
  gemm(hid, fc2yt, fc2y_b, y1, yout, nullptr, M, MLPD, C, 0);
}
